// SelfAttentionBlock_56521769616105
// MI455X (gfx1250) — matrix operand order repaired and checked by reading, not compiled, not run on hardware
//
#include <hip/hip_runtime.h>

// ---------------------------------------------------------------------------
// Fused self-attention block for MI455X (gfx1250, wave32, WMMA + TDM).
// All GEMMs use v_wmma_f32_16x16x32_bf16 (bf16 in, f32 accumulate).
// Operand staging via Tensor Data Mover (tensor_load_to_lds) when available.
// ---------------------------------------------------------------------------

typedef __attribute__((ext_vector_type(4)))  unsigned short ushort4v;
typedef __attribute__((ext_vector_type(8)))  unsigned short ushort8v;
typedef __attribute__((ext_vector_type(8)))  __bf16        v8bf;
typedef __attribute__((ext_vector_type(16))) __bf16        v16bf;
typedef __attribute__((ext_vector_type(8)))  float         v8f;
typedef __attribute__((ext_vector_type(4)))  unsigned int  u32x4;
typedef __attribute__((ext_vector_type(8)))  int           i32x8;
typedef __attribute__((ext_vector_type(4)))  int           i32x4;

#if defined(__gfx1250__) && __has_builtin(__builtin_amdgcn_tensor_load_to_lds) && \
    __has_builtin(__builtin_amdgcn_s_wait_tensorcnt)
#define USE_TDM 1
#endif

__device__ __forceinline__ unsigned short f2bf(float f) {
  unsigned int u = __builtin_bit_cast(unsigned int, f);
  u += 0x7FFFu + ((u >> 16) & 1u);            // round-to-nearest-even
  return (unsigned short)(u >> 16);
}

// ---------------------------------------------------------------------------
// float32 -> bf16 elementwise (4 per thread, vectorized load/store)
// ---------------------------------------------------------------------------
__global__ __launch_bounds__(256) void cvt_bf16_kernel(
    const float* __restrict__ X, unsigned short* __restrict__ Y) {
  size_t idx = ((size_t)blockIdx.x * 256 + threadIdx.x) * 4;
  float4 v = *(const float4*)(X + idx);
  ushort4v o;
  o.x = f2bf(v.x); o.y = f2bf(v.y); o.z = f2bf(v.z); o.w = f2bf(v.w);
  *(ushort4v*)(Y + idx) = o;
}

// W[k][n] (f32) -> Wt[n][k] (bf16) so GEMM B operand has contiguous K per column.
__global__ __launch_bounds__(256) void transpose_w_kernel(
    const float* __restrict__ W, unsigned short* __restrict__ Wt,
    int KD, int ND) {
  int idx = blockIdx.x * 256 + threadIdx.x;   // over ND*KD, output-linear
  int n = idx / KD;
  int k = idx - n * KD;
  Wt[(size_t)n * KD + k] = f2bf(W[(size_t)k * ND + n]);
}

// ---------------------------------------------------------------------------
// Generic bf16 WMMA GEMM:  Out = A[MxK] * Bt[NxK]^T  (+ epilogue by MODE)
//   MODE 0: += bias[n], store bf16 row-major              (Q, K projections)
//   MODE 1: += bias[n], store bf16 transposed [b][n][s]   (V -> Vt)
//   MODE 2: *= scale,   store f32 row-major               (scores)
//   MODE 3: += resid,   store f32 row-major               (context+residual)
// 128 threads = 4 waves; tile 64x64, K-step 32; wave w owns rows 16w..16w+15.
// Double-buffered LDS; staging via TDM (tensor_load_to_lds) when available.
// ---------------------------------------------------------------------------
#define TM 64
#define TN 64
#define TK 32
#define LSTR 40   // LDS row stride in bf16 elements: 80 bytes, 16B-aligned rows

#ifdef USE_TDM
// Issue a TDM 2D tile load: tile [tileRows x TK bf16] from a row-major
// [rows x rowElems] bf16 tensor at gaddr (tile start), into LDS at lds_off,
// with 4-DWORD padding every 16 DWORDs (=> LSTR=40 element LDS row stride).
// D# layout per CDNA5 ISA ch.8 (group0: control/lds/global addr, type=2;
// group1: data_size=2B, pad fields, dims, tile dims, dim0 stride).
__device__ __forceinline__ void tdm_load_2d(const void* gaddr, unsigned lds_off,
                                            int rowElems, int rows,
                                            int tileRows) {
  unsigned long long ga = (unsigned long long)gaddr;
  u32x4 g0;
  g0[0] = 1u;                                         // count=1, user mode
  g0[1] = lds_off;                                    // lds_addr (bytes)
  g0[2] = (unsigned)(ga & 0xFFFFFFFFull);             // global_addr[31:0]
  g0[3] = (unsigned)((ga >> 32) & 0x1FFFFFFull)       // global_addr[56:32]
          | (2u << 30);                               // type = 2 ("image")
  i32x8 g1;
  g1[0] = (int)((1u << 16)                            // data_size = 2 bytes
                | (1u << 20)                          // pad_enable
                | (3u << 22)                          // pad_interval: 16 DWORDs
                | (3u << 25));                        // pad_amount: 4 DWORDs
  g1[1] = (int)(((unsigned)rowElems & 0xFFFFu) << 16);          // dim0[15:0]
  g1[2] = (int)((((unsigned)rowElems >> 16) & 0xFFFFu)          // dim0[31:16]
                | (((unsigned)rows & 0xFFFFu) << 16));          // dim1[15:0]
  g1[3] = (int)((((unsigned)rows >> 16) & 0xFFFFu)              // dim1[31:16]
                | ((unsigned)TK << 16));                        // tile_dim0
  g1[4] = (int)(unsigned)tileRows;                    // tile_dim1 (tile_dim2=0)
  g1[5] = rowElems;                                   // dim0_stride[31:0]
  g1[6] = 0;                                          // dim0_stride hi, dim1_stride lo
  g1[7] = 0;
  i32x4 z4 = {0, 0, 0, 0};
  i32x8 z8 = {0, 0, 0, 0, 0, 0, 0, 0};
  // 6-arg form (clang-23 / therock-10.0 headers): (g0, g1, g2, g3, g4, cpol)
  __builtin_amdgcn_tensor_load_to_lds(g0, g1, z4, z4, z8, 0);
}
#endif

template <int MODE>
__global__ __launch_bounds__(128) void gemm_wmma(
    const unsigned short* __restrict__ A,    // [M x K] bf16 row-major
    const unsigned short* __restrict__ Bt,   // [N x K] bf16 row-major
    void* __restrict__ Out,
    const float* __restrict__ bias,
    const float* __restrict__ resid,
    int M, int N, int K, float scale, int vtS,
    size_t aBS, size_t bBS, size_t oBS, size_t rBS) {
  const int bz = blockIdx.z;
  A  += aBS * (size_t)bz;
  Bt += bBS * (size_t)bz;
  const size_t outBase = oBS * (size_t)bz;
  const size_t resBase = rBS * (size_t)bz;

  __shared__ __align__(16) unsigned short As[2][TM][LSTR];
  __shared__ __align__(16) unsigned short Bs[2][TN][LSTR];

  const int tid  = threadIdx.x;
  const int wave = tid >> 5;
  const int lane = tid & 31;
  const int m0 = blockIdx.y * TM;
  const int n0 = blockIdx.x * TN;
  const int NK = K / TK;

  v8f acc[4] = {};

#ifdef USE_TDM
  // ---- TDM-staged double-buffered pipeline -------------------------------
  const unsigned ldsA[2] = {(unsigned)(unsigned long long)&As[0][0][0],
                            (unsigned)(unsigned long long)&As[1][0][0]};
  const unsigned ldsB[2] = {(unsigned)(unsigned long long)&Bs[0][0][0],
                            (unsigned)(unsigned long long)&Bs[1][0][0]};
  // Wave 0 DMAs the A tile, wave 1 the B tile (TDM ignores EXEC; one issue
  // per wave). Each wave tracks its own TENSORcnt.
  if (wave == 0) tdm_load_2d(A + (size_t)m0 * K, ldsA[0], K, M, TM);
  if (wave == 1) tdm_load_2d(Bt + (size_t)n0 * K, ldsB[0], K, N, TN);
#else
  // ---- manual double-buffered staging ------------------------------------
  const int stageRow = tid >> 1;              // 0..63
  const int stageKh  = (tid & 1) << 4;        // 0 or 16
  const unsigned short* aPtr = A  + (size_t)(m0 + stageRow) * K + stageKh;
  const unsigned short* bPtr = Bt + (size_t)(n0 + stageRow) * K + stageKh;
  {
    ushort8v a0 = *(const ushort8v*)(aPtr);
    ushort8v a1 = *(const ushort8v*)(aPtr + 8);
    ushort8v b0 = *(const ushort8v*)(bPtr);
    ushort8v b1 = *(const ushort8v*)(bPtr + 8);
    *(ushort8v*)&As[0][stageRow][stageKh]     = a0;
    *(ushort8v*)&As[0][stageRow][stageKh + 8] = a1;
    *(ushort8v*)&Bs[0][stageRow][stageKh]     = b0;
    *(ushort8v*)&Bs[0][stageRow][stageKh + 8] = b1;
  }
#endif

  for (int kk = 0; kk < NK; ++kk) {
    const int buf = kk & 1;

#ifdef USE_TDM
    if (kk + 1 < NK) {
      const int k1 = (kk + 1) * TK;
      if (wave == 0) tdm_load_2d(A + (size_t)m0 * K + k1, ldsA[buf ^ 1], K, M, TM);
      if (wave == 1) tdm_load_2d(Bt + (size_t)n0 * K + k1, ldsB[buf ^ 1], K, N, TN);
      if (wave < 2) __builtin_amdgcn_s_wait_tensorcnt(1);  // tile kk landed
    } else {
      if (wave < 2) __builtin_amdgcn_s_wait_tensorcnt(0);
    }
    __syncthreads();   // tile kk visible to all waves
#else
    ushort8v a0, a1, b0, b1;
    if (kk + 1 < NK) {
      const int k1 = (kk + 1) * TK;
      a0 = *(const ushort8v*)(aPtr + k1);
      a1 = *(const ushort8v*)(aPtr + k1 + 8);
      b0 = *(const ushort8v*)(bPtr + k1);
      b1 = *(const ushort8v*)(bPtr + k1 + 8);
      __builtin_prefetch(aPtr + k1 + TK, 0, 1);   // global_prefetch_b8
      __builtin_prefetch(bPtr + k1 + TK, 0, 1);
    }
    __syncthreads();   // tile kk (stored last iter / prologue) visible
#endif

    // A fragment (16x32 bf16): lane<16 -> K0-7 & K16-23; lane>=16 -> K8-15 & K24-31
    const int rowA = (wave << 4) + (lane & 15);
    const int f0   = (lane >> 4) << 3;           // 0 or 8
    const unsigned short* asrc = &As[buf][rowA][0];
    v8bf alo = __builtin_bit_cast(v8bf, *(const ushort8v*)(asrc + f0));
    v8bf ahi = __builtin_bit_cast(v8bf, *(const ushort8v*)(asrc + f0 + 16));
    v16bf af = __builtin_shufflevector(alo, ahi, 0, 1, 2, 3, 4, 5, 6, 7,
                                       8, 9, 10, 11, 12, 13, 14, 15);

    // Load all four B fragments first so the 4 WMMAs can issue back-to-back.
    v16bf bfr[4];
#pragma unroll
    for (int t = 0; t < 4; ++t) {
      // B fragment (32x16 bf16): lane<16 -> K0-7 & K16-23 of col lane; lane>=16 -> K8-15 & K24-31
      const unsigned short* bsrc = &Bs[buf][(t << 4) + (lane & 15)][0];
      const int kb = (lane >> 4) << 3;           // 0 or 8
      v8bf blo = __builtin_bit_cast(v8bf, *(const ushort8v*)(bsrc + kb));
      v8bf bhi = __builtin_bit_cast(v8bf, *(const ushort8v*)(bsrc + kb + 16));
      bfr[t] = __builtin_shufflevector(blo, bhi, 0, 1, 2, 3, 4, 5, 6, 7,
                                       8, 9, 10, 11, 12, 13, 14, 15);
    }
#pragma unroll
    for (int t = 0; t < 4; ++t)
      acc[t] = __builtin_amdgcn_wmma_f32_16x16x32_bf16(
          false, af, false, bfr[t], (short)0, acc[t], false, false);

#ifdef USE_TDM
    __syncthreads();   // all waves done reading buf before TDM refills it
#else
    if (kk + 1 < NK) {
      const int nb = buf ^ 1;
      *(ushort8v*)&As[nb][stageRow][stageKh]     = a0;
      *(ushort8v*)&As[nb][stageRow][stageKh + 8] = a1;
      *(ushort8v*)&Bs[nb][stageRow][stageKh]     = b0;
      *(ushort8v*)&Bs[nb][stageRow][stageKh + 8] = b1;
    }
#endif
  }

  // Epilogue. C layout: VGPR r -> lanes 0-15: (M=r, N=lane); lanes 16-31: (M=r+8, N=lane-16)
  const int mBase = m0 + (wave << 4);
#pragma unroll
  for (int t = 0; t < 4; ++t) {
#pragma unroll
    for (int r = 0; r < 8; ++r) {
      const int mLocal = r + ((lane >> 4) << 3);
      const int nLocal = lane & 15;
      const int row = mBase + mLocal;
      const int col = n0 + (t << 4) + nLocal;
      float v = acc[t][r];
      if constexpr (MODE == 0) {
        v += bias[col];
        ((unsigned short*)Out)[outBase + (size_t)row * N + col] = f2bf(v);
      } else if constexpr (MODE == 1) {
        v += bias[col];
        const int bb = row / vtS;
        const int ss = row - bb * vtS;
        ((unsigned short*)Out)[((size_t)bb * N + col) * (size_t)vtS + ss] = f2bf(v);
      } else if constexpr (MODE == 2) {
        ((float*)Out)[outBase + (size_t)row * N + col] = v * scale;
      } else {
        ((float*)Out)[outBase + (size_t)row * N + col] =
            v + resid[resBase + (size_t)row * N + col];
      }
    }
  }
}

// ---------------------------------------------------------------------------
// Row softmax over 2048 columns (one block per row), bf16 output.
// ---------------------------------------------------------------------------
__global__ __launch_bounds__(256) void softmax_kernel(
    const float* __restrict__ Sc, unsigned short* __restrict__ P, int n) {
  __shared__ float red[8];
  const size_t base = (size_t)blockIdx.x * n;
  const int tid = threadIdx.x;
  float v[8];
  float m = -3.0e38f;
#pragma unroll
  for (int i = 0; i < 8; ++i) {
    v[i] = Sc[base + (size_t)i * 256 + tid];
    m = fmaxf(m, v[i]);
  }
#pragma unroll
  for (int off = 16; off; off >>= 1) m = fmaxf(m, __shfl_xor(m, off, 32));
  if ((tid & 31) == 0) red[tid >> 5] = m;
  __syncthreads();
  float mm = red[0];
#pragma unroll
  for (int i = 1; i < 8; ++i) mm = fmaxf(mm, red[i]);
  __syncthreads();

  float s = 0.0f;
#pragma unroll
  for (int i = 0; i < 8; ++i) {
    v[i] = __expf(v[i] - mm);
    s += v[i];
  }
#pragma unroll
  for (int off = 16; off; off >>= 1) s += __shfl_xor(s, off, 32);
  if ((tid & 31) == 0) red[tid >> 5] = s;
  __syncthreads();
  float ss = 0.0f;
#pragma unroll
  for (int i = 0; i < 8; ++i) ss += red[i];
  const float inv = 1.0f / ss;
#pragma unroll
  for (int i = 0; i < 8; ++i)
    P[base + (size_t)i * 256 + tid] = f2bf(v[i] * inv);
}

// ---------------------------------------------------------------------------
// Row LayerNorm over 1024 columns (one block per row), eps = 1e-3.
// ---------------------------------------------------------------------------
__global__ __launch_bounds__(256) void layernorm_kernel(
    const float* __restrict__ R, const float* __restrict__ g,
    const float* __restrict__ b, float* __restrict__ O) {
  __shared__ float red[16];
  const size_t base = (size_t)blockIdx.x * 1024;
  const int tid = threadIdx.x;
  float v[4];
  float s = 0.0f, s2 = 0.0f;
#pragma unroll
  for (int i = 0; i < 4; ++i) {
    v[i] = R[base + (size_t)i * 256 + tid];
    s += v[i];
    s2 += v[i] * v[i];
  }
#pragma unroll
  for (int off = 16; off; off >>= 1) {
    s  += __shfl_xor(s, off, 32);
    s2 += __shfl_xor(s2, off, 32);
  }
  if ((tid & 31) == 0) {
    red[tid >> 5]       = s;
    red[8 + (tid >> 5)] = s2;
  }
  __syncthreads();
  float ts = 0.0f, ts2 = 0.0f;
#pragma unroll
  for (int i = 0; i < 8; ++i) {
    ts  += red[i];
    ts2 += red[8 + i];
  }
  const float mean = ts * (1.0f / 1024.0f);
  const float var  = ts2 * (1.0f / 1024.0f) - mean * mean;
  const float rstd = rsqrtf(var + 1.0e-3f);
#pragma unroll
  for (int i = 0; i < 4; ++i) {
    const int col = i * 256 + tid;
    O[base + col] = (v[i] - mean) * rstd * g[col] + b[col];
  }
}

// ---------------------------------------------------------------------------
// Host launcher
// ---------------------------------------------------------------------------
extern "C" void kernel_launch(void* const* d_in, const int* in_sizes, int n_in,
                              void* d_out, int out_size, void* d_ws, size_t ws_size,
                              hipStream_t stream) {
  const float* x   = (const float*)d_in[0];
  const float* Wq  = (const float*)d_in[1];
  const float* bq  = (const float*)d_in[2];
  const float* Wk  = (const float*)d_in[3];
  const float* bk  = (const float*)d_in[4];
  const float* Wv  = (const float*)d_in[5];
  const float* bv  = (const float*)d_in[6];
  const float* gam = (const float*)d_in[7];
  const float* bet = (const float*)d_in[8];
  float* out = (float*)d_out;

  const int Bn = 4, S = 2048, D = 1024;
  const int Ntok = Bn * S;

  // workspace layout (bytes; all offsets 2MB-multiples -> well aligned)
  char* ws = (char*)d_ws;
  size_t off = 0;
  unsigned short* Xb  = (unsigned short*)(ws + off); off += (size_t)Ntok * D * 2;  // 16MB
  unsigned short* Wtq = (unsigned short*)(ws + off); off += (size_t)D * D * 2;     //  2MB
  unsigned short* Wtk = (unsigned short*)(ws + off); off += (size_t)D * D * 2;
  unsigned short* Wtv = (unsigned short*)(ws + off); off += (size_t)D * D * 2;
  unsigned short* Qb  = (unsigned short*)(ws + off); off += (size_t)Ntok * D * 2;  // 16MB
  unsigned short* Kb  = (unsigned short*)(ws + off); off += (size_t)Ntok * D * 2;  // 16MB
  unsigned short* Vt  = (unsigned short*)(ws + off); off += (size_t)Ntok * D * 2;  // 16MB [B][D][S]
  float*          Sc  = (float*)(ws + off);          off += (size_t)Bn * S * S * 4; // 64MB
  unsigned short* P   = (unsigned short*)(ws + off); off += (size_t)Bn * S * S * 2; // 32MB
  float*          Rs  = (float*)(ws + off);          off += (size_t)Ntok * D * 4;  // 32MB

  // 1) precision conversion / weight transpose
  cvt_bf16_kernel<<<(Ntok * D) / 1024, 256, 0, stream>>>(x, Xb);
  transpose_w_kernel<<<(D * D) / 256, 256, 0, stream>>>(Wq, Wtq, D, D);
  transpose_w_kernel<<<(D * D) / 256, 256, 0, stream>>>(Wk, Wtk, D, D);
  transpose_w_kernel<<<(D * D) / 256, 256, 0, stream>>>(Wv, Wtv, D, D);

  // 2) QKV projections: [8192x1024] x [1024x1024]
  dim3 gQKV(D / TN, Ntok / TM, 1);
  gemm_wmma<0><<<gQKV, 128, 0, stream>>>(Xb, Wtq, Qb, bq, nullptr,
                                         Ntok, D, D, 1.0f, S, 0, 0, 0, 0);
  gemm_wmma<0><<<gQKV, 128, 0, stream>>>(Xb, Wtk, Kb, bk, nullptr,
                                         Ntok, D, D, 1.0f, S, 0, 0, 0, 0);
  gemm_wmma<1><<<gQKV, 128, 0, stream>>>(Xb, Wtv, Vt, bv, nullptr,
                                         Ntok, D, D, 1.0f, S, 0, 0, 0, 0);

  // 3) scores = Q @ K^T / sqrt(D)   per batch (K stored row-major == Bt layout)
  dim3 gSc(S / TN, S / TM, Bn);
  gemm_wmma<2><<<gSc, 128, 0, stream>>>(Qb, Kb, Sc, nullptr, nullptr,
                                        S, S, D, 1.0f / 32.0f, S,
                                        (size_t)S * D, (size_t)S * D,
                                        (size_t)S * S, 0);

  // 4) row softmax -> P (bf16)
  softmax_kernel<<<Ntok, 256, 0, stream>>>(Sc, P, S);

  // 5) context = P @ V  (+ residual), Vt is [B][D][S] == Bt layout per batch
  dim3 gCtx(D / TN, S / TM, Bn);
  gemm_wmma<3><<<gCtx, 128, 0, stream>>>(P, Vt, Rs, nullptr, x,
                                         S, D, S, 1.0f, S,
                                         (size_t)S * S, (size_t)D * S,
                                         (size_t)S * D, (size_t)S * D);

  // 6) LayerNorm -> output
  layernorm_kernel<<<Ntok, 256, 0, stream>>>(Rs, gam, bet, out);
}
